// SelfAttention_48739288875429
// MI455X (gfx1250) — matrix operand order repaired and checked by reading, not compiled, not run on hardware
//
#include <hip/hip_runtime.h>
#include <hip/hip_bf16.h>
#include <math.h>

// B=16, T1=512, T2=64, D=768, H=12, E=64, TK=T1+T2=576
#define DD 768
#define TKK 576
#define PROJ_SCALE 0.35355339059327373f  // 64^-0.25

typedef __attribute__((ext_vector_type(16))) _Float16 v16h;
typedef __attribute__((ext_vector_type(8)))  float    v8f;
typedef __attribute__((ext_vector_type(4)))  unsigned int u32x4;
typedef int v4i __attribute__((vector_size(16)));  // matches builtin param type

union FragU { u32x4 u[2]; v16h v; };

#if defined(__has_builtin)
#if __has_builtin(__builtin_amdgcn_global_load_async_to_lds_b128)
#define HAVE_ASYNC_LDS 1
#endif
#endif

// stage 16 bytes global -> LDS (async on CDNA5, sync fallback otherwise)
__device__ __forceinline__ void stage16(const _Float16* g, _Float16* l) {
#ifdef HAVE_ASYNC_LDS
  __builtin_amdgcn_global_load_async_to_lds_b128(
      (__attribute__((address_space(1))) v4i*)g,
      (__attribute__((address_space(3))) v4i*)l, 0, 0);
#else
  *(u32x4*)l = *(const u32x4*)g;
#endif
}

__device__ __forceinline__ void wait_async() {
#ifdef HAVE_ASYNC_LDS
  asm volatile("s_wait_asynccnt 0" ::: "memory");
#endif
}

// ---------------------------------------------------------------------------
// elementwise f32 -> f16
__global__ void cvt_f16_kernel(const float* __restrict__ in,
                               _Float16* __restrict__ out, int n) {
  int i = blockIdx.x * blockDim.x + threadIdx.x;
  if (i < n) out[i] = (_Float16)in[i];
}

// W [nmat,768,768] f32 -> Wt [nmat,768,768] f16 with Wt[m][f][d] = W[m][d][f]
__global__ void transpose_cvt_kernel(const float* __restrict__ W,
                                     _Float16* __restrict__ Wt, int n) {
  int i = blockIdx.x * blockDim.x + threadIdx.x;
  if (i >= n) return;
  int m = i / (DD * DD);
  int rem = i - m * DD * DD;
  int f = rem / DD;
  int d = rem - f * DD;
  Wt[i] = (_Float16)W[(size_t)m * DD * DD + (size_t)d * DD + f];
}

// ---------------------------------------------------------------------------
// GEMM: out = (A[M,768] @ Wt^T + bias) * scale  (Wt stored [N=768, K=768])
// 128 threads = 4 waves; block tile 128 rows x 64 cols; wave owns 32 rows
// (two 16-row subtiles sharing each B fragment). B tile async-staged in LDS.
// mode 0: f16, row = b*outT + key_off + t      (Q/K buffers incl. key concat)
// mode 1: f16 transposed, idx = (b*768+col)*576 + key_off + t   (V^T buffers)
// mode 2: f32 row-major [M,768]                 (final outputs)
__global__ __launch_bounds__(128) void gemm_f16_kernel(
    const _Float16* __restrict__ A, const _Float16* __restrict__ Wt,
    const float* __restrict__ bias, void* __restrict__ outp,
    int T, int outT, int key_off, float scale, int mode) {
  __shared__ __align__(16) _Float16 sB[64 * 32];   // 4 KB B tile
  const int tid  = threadIdx.x;
  const int lane = tid & 31;
  const int wave = tid >> 5;
  const int row0 = blockIdx.x * 128 + wave * 32;
  const int col0 = blockIdx.y * 64;
  const int nn_l = lane & 15;
  const int ab  = (lane < 16) ? 0 : 8;    // A fragment k-base
  const int kb  = (lane < 16) ? 0 : 8;    // B fragment k-base
  const int hi8 = (lane < 16) ? 0 : 8;    // C row offset
  const int srow  = tid >> 1;             // staging row 0..63
  const int spart = tid & 1;              // staging half (16 elems each)

  v8f acc[2][4] = {};
  const _Float16* Ar0  = A + (size_t)(row0 + nn_l) * DD;
  const _Float16* Ar1  = A + (size_t)(row0 + 16 + nn_l) * DD;
  const _Float16* Wrow = Wt + (size_t)(col0 + srow) * DD;

  for (int k0 = 0; k0 < DD; k0 += 32) {
    // stage B tile [64 x 32] cooperatively (2 x b128 per thread)
    {
      const _Float16* g = Wrow + k0 + spart * 16;
      _Float16* l = sB + srow * 32 + spart * 16;
      stage16(g, l);
      stage16(g + 8, l + 8);
    }
    FragU a0, a1;
    const _Float16* pa0 = Ar0 + k0 + ab;
    a0.u[0] = *(const u32x4*)pa0;
    a0.u[1] = *(const u32x4*)(pa0 + 16);
    const _Float16* pa1 = Ar1 + k0 + ab;
    a1.u[0] = *(const u32x4*)pa1;
    a1.u[1] = *(const u32x4*)(pa1 + 16);
    if (k0 + 32 < DD) __builtin_prefetch(pa0 + 32, 0, 1);
    wait_async();
    __syncthreads();
#pragma unroll
    for (int nt = 0; nt < 4; ++nt) {
      FragU bf;
      const _Float16* pb = sB + (nt * 16 + nn_l) * 32 + kb;
      bf.u[0] = *(const u32x4*)pb;
      bf.u[1] = *(const u32x4*)(pb + 16);
      acc[0][nt] = __builtin_amdgcn_wmma_f32_16x16x32_f16(
          false, a0.v, false, bf.v, (short)0, acc[0][nt], false, false);
      acc[1][nt] = __builtin_amdgcn_wmma_f32_16x16x32_f16(
          false, a1.v, false, bf.v, (short)0, acc[1][nt], false, false);
    }
    __syncthreads();  // before next stage overwrites sB
  }

#pragma unroll
  for (int sub = 0; sub < 2; ++sub) {
#pragma unroll
    for (int nt = 0; nt < 4; ++nt) {
      const int nn = col0 + nt * 16 + nn_l;
      const float bv = bias[nn];
#pragma unroll
      for (int r = 0; r < 8; ++r) {
        int mm = row0 + sub * 16 + r + hi8;
        float val = (acc[sub][nt][r] + bv) * scale;
        if (mode == 2) {
          ((float*)outp)[(size_t)mm * DD + nn] = val;
        } else {
          int bb = mm / T, tt = mm - bb * T;
          _Float16 hv = (_Float16)val;
          if (mode == 0)
            ((_Float16*)outp)[(size_t)(bb * outT + key_off + tt) * DD + nn] = hv;
          else
            ((_Float16*)outp)[((size_t)bb * DD + nn) * TKK + key_off + tt] = hv;
        }
      }
    }
  }
}

// ---------------------------------------------------------------------------
// Fused attention: one wave32 per (b, h, 16-query tile).
// scores = [Q1|Q2-tile] @ K^T -> LDS (f16), mask+softmax in place, P @ V.
// Q1 used for key cols < 512 (visual keys), Q2 for key cols >= 512.
// Kb: [B,576,768] f16 row-major; VT: [B,768,576] f16 (values transposed).
// LDS: 16x576 f16 = 18.4 KB -> high WGP occupancy.
__global__ __launch_bounds__(32) void attn_kernel(
    const _Float16* __restrict__ Q1, const _Float16* __restrict__ Q2,
    const _Float16* __restrict__ Kb, const _Float16* __restrict__ VT,
    const unsigned char* __restrict__ mask, _Float16* __restrict__ O, int Tq) {
  __shared__ __align__(16) _Float16 sS[16 * TKK];
  const int lane = threadIdx.x;
  const int qt = blockIdx.x, h = blockIdx.y, b = blockIdx.z;
  const int q0 = qt * 16;
  const int nn_l = lane & 15;
  const int ab  = (lane < 16) ? 0 : 8;
  const int kb  = (lane < 16) ? 0 : 8;
  const int hi8 = (lane < 16) ? 0 : 8;

  // ---- scores: 36 key tiles x (K=64 -> 2 wmma steps) ----
  for (int kt = 0; kt < 36; ++kt) {
    int n0 = kt * 16;
    const _Float16* Q = (n0 < 512) ? Q1 : Q2;
    const _Float16* Arow = Q + ((size_t)b * Tq + q0 + nn_l) * DD + h * 64;
    const _Float16* Krow = Kb + ((size_t)b * TKK + n0 + nn_l) * DD + h * 64;
    v8f acc = {};
#pragma unroll
    for (int ks = 0; ks < 64; ks += 32) {
      FragU a, bf;
      const _Float16* pa = Arow + ks + ab;
      a.u[0] = *(const u32x4*)pa;
      a.u[1] = *(const u32x4*)(pa + 16);
      const _Float16* pb = Krow + ks + kb;
      bf.u[0] = *(const u32x4*)pb;
      bf.u[1] = *(const u32x4*)(pb + 16);
      acc = __builtin_amdgcn_wmma_f32_16x16x32_f16(
          false, a.v, false, bf.v, (short)0, acc, false, false);
    }
#pragma unroll
    for (int r = 0; r < 8; ++r)
      sS[(r + hi8) * TKK + n0 + nn_l] = (_Float16)acc[r];
  }
  __syncthreads();

  // ---- mask + softmax in place (2 lanes per row, halves via shfl_xor 16) --
  {
    const int row = lane & 15;
    const int half = lane >> 4;
    const int c0 = half * 288;
    _Float16* srow = sS + row * TKK + c0;
    const unsigned char* mrow = mask + (size_t)b * TKK + c0;
    float mx = -3.0e38f;
    for (int c = 0; c < 288; ++c) {
      float s = mrow[c] ? -60000.0f : (float)srow[c];
      mx = fmaxf(mx, s);
    }
    mx = fmaxf(mx, __shfl_xor(mx, 16, 32));
    float sum = 0.f;
    for (int c = 0; c < 288; ++c) {
      float s = mrow[c] ? -60000.0f : (float)srow[c];
      float e = __expf(s - mx);
      srow[c] = (_Float16)e;
      sum += e;
    }
    sum += __shfl_xor(sum, 16, 32);
    float inv = 1.0f / sum;
    for (int c = 0; c < 288; ++c)
      srow[c] = (_Float16)((float)srow[c] * inv);
  }
  __syncthreads();

  // ---- out = P @ V : 4 col tiles x 18 K-chunks; A-frags from LDS ----
#pragma unroll
  for (int nt = 0; nt < 4; ++nt) {
    v8f acc = {};
    const _Float16* Vrow =
        VT + ((size_t)b * DD + h * 64 + nt * 16 + nn_l) * TKK;
    for (int kc = 0; kc < TKK; kc += 32) {
      FragU a, bf;
      const _Float16* pa = sS + nn_l * TKK + kc + ab;
      a.u[0] = *(const u32x4*)pa;
      a.u[1] = *(const u32x4*)(pa + 16);
      const _Float16* pb = Vrow + kc + kb;
      bf.u[0] = *(const u32x4*)pb;
      bf.u[1] = *(const u32x4*)(pb + 16);
      acc = __builtin_amdgcn_wmma_f32_16x16x32_f16(
          false, a.v, false, bf.v, (short)0, acc, false, false);
    }
#pragma unroll
    for (int r = 0; r < 8; ++r) {
      int mm = q0 + r + hi8;
      O[((size_t)b * Tq + mm) * DD + h * 64 + nt * 16 + nn_l] = (_Float16)acc[r];
    }
  }
}

// ---------------------------------------------------------------------------
extern "C" void kernel_launch(void* const* d_in, const int* in_sizes, int n_in,
                              void* d_out, int out_size, void* d_ws, size_t ws_size,
                              hipStream_t stream) {
  (void)in_sizes; (void)n_in; (void)out_size; (void)ws_size;
  const float* feats = (const float*)d_in[0];
  const float* inps  = (const float*)d_in[1];
  const unsigned char* mask = (const unsigned char*)d_in[2];
  const float* W_f = (const float*)d_in[3];
  const float* b_f = (const float*)d_in[4];
  const float* W_i = (const float*)d_in[5];
  const float* b_i = (const float*)d_in[6];
  const float* Wu_v = (const float*)d_in[7];
  const float* bu_v = (const float*)d_in[8];
  const float* Wu_l = (const float*)d_in[9];
  const float* bu_l = (const float*)d_in[10];

  const size_t N_FE = (size_t)16 * 512 * 768;  // 6291456
  const size_t N_IN = (size_t)16 * 64 * 768;   // 786432
  const size_t N_W6 = (size_t)6 * 768 * 768;   // 3538944
  const size_t N_W1 = (size_t)768 * 768;       // 589824
  const size_t N_KV = (size_t)16 * 576 * 768;  // 7077888

  _Float16* ws = (_Float16*)d_ws;
  size_t off = 0;
  _Float16* feats_h = ws + off; off += N_FE;
  _Float16* inps_h  = ws + off; off += N_IN;
  _Float16* Wf_t  = ws + off; off += N_W6;
  _Float16* Wi_t  = ws + off; off += N_W6;
  _Float16* Wuv_t = ws + off; off += N_W1;
  _Float16* Wul_t = ws + off; off += N_W1;
  _Float16* Qvv = ws + off; off += N_FE;
  _Float16* Qlv = ws + off; off += N_FE;
  _Float16* Qvl = ws + off; off += N_IN;
  _Float16* Qll = ws + off; off += N_IN;
  _Float16* Kv  = ws + off; off += N_KV;
  _Float16* Kl  = ws + off; off += N_KV;
  _Float16* vTv = ws + off; off += N_KV;
  _Float16* vTl = ws + off; off += N_KV;
  _Float16* Ov  = ws + off; off += N_FE;
  _Float16* Ol  = ws + off; off += N_IN;

  dim3 blk(256);
  cvt_f16_kernel<<<dim3((unsigned)((N_FE + 255) / 256)), blk, 0, stream>>>(feats, feats_h, (int)N_FE);
  cvt_f16_kernel<<<dim3((unsigned)((N_IN + 255) / 256)), blk, 0, stream>>>(inps, inps_h, (int)N_IN);
  transpose_cvt_kernel<<<dim3((unsigned)((N_W6 + 255) / 256)), blk, 0, stream>>>(W_f, Wf_t, (int)N_W6);
  transpose_cvt_kernel<<<dim3((unsigned)((N_W6 + 255) / 256)), blk, 0, stream>>>(W_i, Wi_t, (int)N_W6);
  transpose_cvt_kernel<<<dim3((unsigned)((N_W1 + 255) / 256)), blk, 0, stream>>>(Wu_v, Wuv_t, (int)N_W1);
  transpose_cvt_kernel<<<dim3((unsigned)((N_W1 + 255) / 256)), blk, 0, stream>>>(Wu_l, Wul_t, (int)N_W1);

  const float s = PROJ_SCALE;
  dim3 gb(64, 12), bb(128);  // M=8192: 64 tiles of 128 rows
  dim3 gi(8, 12);            // M=1024: 8 tiles of 128 rows
  // feats projections: [k_vv, q_vv, v_vv, q_lv, k_vl, v_vl]
  gemm_f16_kernel<<<gb, bb, 0, stream>>>(feats_h, Wf_t + 0 * N_W1, b_f + 0 * 768, Kv,  512, 576, 0, s, 0);
  gemm_f16_kernel<<<gb, bb, 0, stream>>>(feats_h, Wf_t + 1 * N_W1, b_f + 1 * 768, Qvv, 512, 512, 0, s, 0);
  gemm_f16_kernel<<<gb, bb, 0, stream>>>(feats_h, Wf_t + 2 * N_W1, b_f + 2 * 768, vTv, 512, 0,   0, s, 1);
  gemm_f16_kernel<<<gb, bb, 0, stream>>>(feats_h, Wf_t + 3 * N_W1, b_f + 3 * 768, Qlv, 512, 512, 0, s, 0);
  gemm_f16_kernel<<<gb, bb, 0, stream>>>(feats_h, Wf_t + 4 * N_W1, b_f + 4 * 768, Kl,  512, 576, 0, s, 0);
  gemm_f16_kernel<<<gb, bb, 0, stream>>>(feats_h, Wf_t + 5 * N_W1, b_f + 5 * 768, vTl, 512, 0,   0, s, 1);
  // inps projections: [k_lv, v_lv, q_vl, q_ll, k_ll, v_ll]
  gemm_f16_kernel<<<gi, bb, 0, stream>>>(inps_h, Wi_t + 0 * N_W1, b_i + 0 * 768, Kv,  64, 576, 512, s, 0);
  gemm_f16_kernel<<<gi, bb, 0, stream>>>(inps_h, Wi_t + 1 * N_W1, b_i + 1 * 768, vTv, 64, 0,   512, s, 1);
  gemm_f16_kernel<<<gi, bb, 0, stream>>>(inps_h, Wi_t + 2 * N_W1, b_i + 2 * 768, Qvl, 64, 64,  0,   s, 0);
  gemm_f16_kernel<<<gi, bb, 0, stream>>>(inps_h, Wi_t + 3 * N_W1, b_i + 3 * 768, Qll, 64, 64,  0,   s, 0);
  gemm_f16_kernel<<<gi, bb, 0, stream>>>(inps_h, Wi_t + 4 * N_W1, b_i + 4 * 768, Kl,  64, 576, 512, s, 0);
  gemm_f16_kernel<<<gi, bb, 0, stream>>>(inps_h, Wi_t + 5 * N_W1, b_i + 5 * 768, vTl, 64, 0,   512, s, 1);

  // fused attention (one wave32 per b,h,16-query tile)
  attn_kernel<<<dim3(32, 12, 16), dim3(32), 0, stream>>>(Qvv, Qlv, Kv, vTv, mask, Ov, 512);
  attn_kernel<<<dim3(4, 12, 16),  dim3(32), 0, stream>>>(Qvl, Qll, Kl, vTl, mask, Ol, 64);

  // output projections, f32 straight into d_out
  float* outv = (float*)d_out;
  float* outl = outv + N_FE;
  gemm_f16_kernel<<<gb, bb, 0, stream>>>(Ov, Wuv_t, bu_v, outv, 512, 0, 0, 1.0f, 2);
  gemm_f16_kernel<<<gi, bb, 0, stream>>>(Ol, Wul_t, bu_l, outl, 64, 0, 0, 1.0f, 2);
}
